// FP8Experts_54760833024692
// MI455X (gfx1250) — 8-bit matrix operand order repaired and checked by reading, compiled, not run on hardware
//
#include <hip/hip_runtime.h>
#include <stdint.h>

#define E_      8
#define H_      2048
#define I_      1408
#define T_      4096
#define TOPK_   2
#define BK_     128
#define FP8MAX  448.0f
#define GU_N    (2 * I_)
#define GU_NB   (GU_N / BK_)
#define H_KB    (H_ / BK_)
#define I_KB    (I_ / BK_)
#define H_NB    (H_ / BK_)
#define MTILE   64
#define NSUB    (MTILE / 16)
#define NROW    (T_ * TOPK_ + E_ * MTILE)
#define NTILE_MAX (NROW / MTILE)
#define ASTRIDE1 2064
#define ASTRIDE2 1424

typedef int   v16i __attribute__((ext_vector_type(16)));
typedef float v8f  __attribute__((ext_vector_type(8)));

constexpr size_t OFF_XQ    = 0;
constexpr size_t OFF_XS    = OFF_XQ   + (size_t)T_ * H_;
constexpr size_t OFF_IQ    = OFF_XS   + (size_t)T_ * H_KB * 4;
constexpr size_t OFF_IS    = OFF_IQ   + (size_t)NROW * I_;
constexpr size_t OFF_RTOK  = OFF_IS   + (size_t)NROW * I_KB * 4;
constexpr size_t OFF_RW    = OFF_RTOK + (size_t)NROW * 4;
constexpr size_t OFF_CNT   = OFF_RW   + (size_t)NROW * 4;
constexpr size_t OFF_CUR   = OFF_CNT  + 64;
constexpr size_t OFF_OFFS  = OFF_CUR  + 64;
constexpr size_t OFF_TILEE = OFF_OFFS + 64;
constexpr size_t OFF_TILER = OFF_TILEE + NTILE_MAX * 4;
constexpr size_t OFF_NT    = OFF_TILER + NTILE_MAX * 4;

__device__ __forceinline__ float clamp448(float v) {
    return fminf(fmaxf(v, -FP8MAX), FP8MAX);
}

__device__ __forceinline__ unsigned sw_e4m3(float x) {
    unsigned u = __float_as_uint(x);
    unsigned s = (u >> 24) & 0x80u;
    float a = fabsf(x);
    if (a >= 448.f) return s | 0x7Eu;
    if (a < 0.015625f) {
        int m = (int)rintf(a * 512.f);
        return s | (unsigned)m;
    }
    unsigned au = __float_as_uint(a);
    unsigned r  = au + 0x0007FFFFu + ((au >> 20) & 1u);
    unsigned e4 = (r >> 23) & 0xFFu;
    if (e4 < 121u) return s;
    unsigned ee = e4 - 120u;
    unsigned mm = (r >> 20) & 7u;
    if (ee > 15u || (ee == 15u && mm == 7u)) return s | 0x7Eu;
    return s | (ee << 4) | mm;
}

__device__ __forceinline__ unsigned pack_fp8x4(float a, float b, float c, float d) {
#if __has_builtin(__builtin_amdgcn_cvt_pk_fp8_f32)
    int v = 0;
    v = __builtin_amdgcn_cvt_pk_fp8_f32(a, b, v, false);
    v = __builtin_amdgcn_cvt_pk_fp8_f32(c, d, v, true);
    return (unsigned)v;
#else
    return sw_e4m3(a) | (sw_e4m3(b) << 8) | (sw_e4m3(c) << 16) | (sw_e4m3(d) << 24);
#endif
}

__global__ void fe_zero_out_k(float* __restrict__ out, int n4) {
    int i = blockIdx.x * blockDim.x + threadIdx.x;
    if (i < n4) ((float4*)out)[i] = make_float4(0.f, 0.f, 0.f, 0.f);
}

__global__ void fe_init_route_k(int* rowTok, float* rowW, int* cnt, int* cur) {
    int i = blockIdx.x * blockDim.x + threadIdx.x;
    if (i < NROW) { rowTok[i] = 0; rowW[i] = 0.f; }
    if (i < E_)   { cnt[i] = 0; cur[i] = 0; }
}

__global__ void fe_count_k(const int* __restrict__ idx, int* cnt) {
    int i = blockIdx.x * blockDim.x + threadIdx.x;
    if (i < T_ * TOPK_) atomicAdd(&cnt[idx[i]], 1);
}

__global__ void fe_plan_k(const int* __restrict__ cnt, int* offs,
                          int* tileE, int* tileRow, int* nTiles) {
    if (blockIdx.x == 0 && threadIdx.x == 0) {
        int o = 0, tn = 0;
        for (int e = 0; e < E_; ++e) {
            offs[e] = o;
            int nt = (cnt[e] + MTILE - 1) / MTILE;
            for (int i = 0; i < nt; ++i) { tileE[tn] = e; tileRow[tn] = o + i * MTILE; ++tn; }
            o += nt * MTILE;
        }
        offs[E_] = o;
        nTiles[0] = tn;
    }
}

__global__ void fe_scatter_k(const int* __restrict__ idx, const float* __restrict__ w,
                             const int* __restrict__ offs, int* cur,
                             int* rowTok, float* rowW) {
    int i = blockIdx.x * blockDim.x + threadIdx.x;
    if (i < T_ * TOPK_) {
        int e = idx[i];
        int slot = atomicAdd(&cur[e], 1);
        int row = offs[e] + slot;
        rowTok[row] = i / TOPK_;
        rowW[row]   = w[i];
    }
}

__global__ __launch_bounds__(256) void fe_act_quant_k(const float* __restrict__ x,
                                                      uint8_t* __restrict__ xq,
                                                      float* __restrict__ xs) {
    __shared__ float red[256];
    __shared__ float scl[16];
    int t = blockIdx.x, tid = threadIdx.x;
    const float* row = x + (size_t)t * H_;
    float4 p0 = ((const float4*)row)[tid * 2];
    float4 p1 = ((const float4*)row)[tid * 2 + 1];
    float v[8] = {p0.x, p0.y, p0.z, p0.w, p1.x, p1.y, p1.z, p1.w};
    float am = 0.f;
#pragma unroll
    for (int j = 0; j < 8; ++j) am = fmaxf(am, fabsf(v[j]));
    red[tid] = am;
    __syncthreads();
    if (tid < 16) {
        float m = 0.f;
#pragma unroll
        for (int j = 0; j < 16; ++j) m = fmaxf(m, red[tid * 16 + j]);
        float s = fmaxf(m, 1e-12f) / FP8MAX;
        scl[tid] = s;
        xs[t * H_KB + tid] = s;
    }
    __syncthreads();
    float inv = 1.f / scl[tid >> 4];
    float q[8];
#pragma unroll
    for (int j = 0; j < 8; ++j) q[j] = clamp448(v[j] * inv);
    uint2 o;
    o.x = pack_fp8x4(q[0], q[1], q[2], q[3]);
    o.y = pack_fp8x4(q[4], q[5], q[6], q[7]);
    *(uint2*)(xq + (size_t)t * H_ + tid * 8) = o;
}

__global__ __launch_bounds__(256) void fe_gemm1_k(
    const uint8_t* __restrict__ xq, const float* __restrict__ xs,
    const uint8_t* __restrict__ wgu, const float* __restrict__ gu_sinv,
    const int* __restrict__ rowTok,
    const int* __restrict__ tileE, const int* __restrict__ tileRow,
    const int* __restrict__ nTiles,
    uint8_t* __restrict__ iq, float* __restrict__ isc) {
    __shared__ __align__(16) uint8_t smem[MTILE * ASTRIDE1];
    __shared__ int   tokLds[MTILE];
    __shared__ float xsLds[MTILE * H_KB];
    __shared__ float sclLds[MTILE];
    uint8_t* Alds     = smem;
    float*   interLds = (float*)smem;
    float*   redLds   = (float*)(smem + MTILE * 128 * 4);

    int tile = blockIdx.y;
    if (tile >= nTiles[0]) return;
    int e = tileE[tile], rowBase = tileRow[tile];
    int tid = threadIdx.x;

    if (tid < MTILE) tokLds[tid] = rowTok[rowBase + tid];
    __syncthreads();
#pragma unroll
    for (int it = 0; it < 4; ++it) {
        int idx = tid + it * 256;
        xsLds[idx] = xs[tokLds[idx >> 4] * H_KB + (idx & 15)];
    }
#pragma unroll
    for (int it = 0; it < 32; ++it) {
        int idx = tid + it * 256;
        int row = idx >> 7, chunk = idx & 127;
        *(uint4*)(Alds + row * ASTRIDE1 + chunk * 16) =
            *(const uint4*)(xq + (size_t)tokLds[row] * H_ + chunk * 16);
    }
    __syncthreads();

    int lane = tid & 31, wave = tid >> 5;
    int nIn = lane & 15, khalf = lane >> 4;
    int gcol = blockIdx.x * 128 + wave * 16 + nIn;
    const uint8_t* wg = wgu + ((size_t)e * GU_N + gcol) * H_;
    const uint8_t* wu = wg + (size_t)I_ * H_;
    const float* sfG = gu_sinv + ((size_t)e * GU_NB + blockIdx.x) * H_KB;
    const float* sfU = gu_sinv + ((size_t)e * GU_NB + I_KB + blockIdx.x) * H_KB;

    float accG[NSUB][8] = {{0}}, accU[NSUB][8] = {{0}};

    for (int kb = 0; kb < H_KB; ++kb) {
        v16i bg, bu;
#pragma unroll
        for (int g = 0; g < 4; ++g) {
            int ko = kb * 128 + g * 32 + khalf * 16;
            uint4 qg = *(const uint4*)(wg + ko);
            uint4 qu = *(const uint4*)(wu + ko);
            bg[4 * g] = (int)qg.x; bg[4 * g + 1] = (int)qg.y;
            bg[4 * g + 2] = (int)qg.z; bg[4 * g + 3] = (int)qg.w;
            bu[4 * g] = (int)qu.x; bu[4 * g + 1] = (int)qu.y;
            bu[4 * g + 2] = (int)qu.z; bu[4 * g + 3] = (int)qu.w;
        }
        float sg = sfG[kb], su = sfU[kb];
#pragma unroll
        for (int s = 0; s < NSUB; ++s) {
            const uint8_t* arow = Alds + (size_t)(s * 16 + (lane & 15)) * ASTRIDE1;
            v16i a;
#pragma unroll
            for (int h = 0; h < 2; ++h)
#pragma unroll
                for (int p = 0; p < 4; ++p) {
                    uint2 d = *(const uint2*)(arow + kb * 128 + h * 64 + (p >> 1) * 32 + khalf * 16 + (p & 1) * 8);
                    a[8 * h + 2 * p]     = (int)d.x;
                    a[8 * h + 2 * p + 1] = (int)d.y;
                }
            v8f zero = {};
            v8f pg = __builtin_amdgcn_wmma_f32_16x16x128_fp8_fp8(a, bg, (short)0, zero, false, false);
            v8f pu = __builtin_amdgcn_wmma_f32_16x16x128_fp8_fp8(a, bu, (short)0, zero, false, false);
            int mBase = s * 16 + khalf * 8;
#pragma unroll
            for (int r = 0; r < 8; ++r) {
                float sa = xsLds[(mBase + r) * H_KB + kb];
                accG[s][r] += pg[r] * (sa * sg);
                accU[s][r] += pu[r] * (sa * su);
            }
        }
    }

    __syncthreads();
#pragma unroll
    for (int s = 0; s < NSUB; ++s) {
        int mBase = s * 16 + khalf * 8;
#pragma unroll
        for (int r = 0; r < 8; ++r) {
            float g = accG[s][r];
            float val = (g / (1.f + __expf(-g))) * accU[s][r];
            interLds[(mBase + r) * 128 + wave * 16 + nIn] = val;
        }
    }
    __syncthreads();
    { int row = tid >> 2, seg = tid & 3;
      float mx = 0.f;
#pragma unroll
      for (int j = 0; j < 32; ++j) mx = fmaxf(mx, fabsf(interLds[row * 128 + seg * 32 + j]));
      redLds[tid] = mx; }
    __syncthreads();
    if (tid < MTILE) {
        float mx = fmaxf(fmaxf(redLds[tid * 4], redLds[tid * 4 + 1]),
                         fmaxf(redLds[tid * 4 + 2], redLds[tid * 4 + 3]));
        float s = fmaxf(mx, 1e-12f) / FP8MAX;
        sclLds[tid] = s;
        isc[(size_t)(rowBase + tid) * I_KB + blockIdx.x] = s;
    }
    __syncthreads();
#pragma unroll
    for (int it = 0; it < 4; ++it) {
        int idx = (tid + it * 256) * 8;
        int row = idx >> 7, c0 = idx & 127;
        float inv = 1.f / sclLds[row];
        float q[8];
#pragma unroll
        for (int j = 0; j < 8; ++j) q[j] = clamp448(interLds[row * 128 + c0 + j] * inv);
        uint2 o;
        o.x = pack_fp8x4(q[0], q[1], q[2], q[3]);
        o.y = pack_fp8x4(q[4], q[5], q[6], q[7]);
        *(uint2*)(iq + (size_t)(rowBase + row) * I_ + blockIdx.x * 128 + c0) = o;
    }
}

__global__ __launch_bounds__(256) void fe_gemm2_k(
    const uint8_t* __restrict__ iq, const float* __restrict__ isc,
    const uint8_t* __restrict__ wd, const float* __restrict__ d_sinv,
    const int* __restrict__ rowTok, const float* __restrict__ rowW,
    const int* __restrict__ tileE, const int* __restrict__ tileRow,
    const int* __restrict__ nTiles, float* __restrict__ out) {
    __shared__ __align__(16) uint8_t Alds[MTILE * ASTRIDE2];
    __shared__ float isLds[MTILE * I_KB];
    __shared__ float wLds[MTILE];
    __shared__ int   tokLds[MTILE];

    int tile = blockIdx.y;
    if (tile >= nTiles[0]) return;
    int e = tileE[tile], rowBase = tileRow[tile];
    int tid = threadIdx.x;

    if (tid < MTILE) { tokLds[tid] = rowTok[rowBase + tid]; wLds[tid] = rowW[rowBase + tid]; }
#pragma unroll
    for (int it = 0; it < 3; ++it) {
        int idx = tid + it * 256;
        if (idx < MTILE * I_KB)
            isLds[idx] = isc[(size_t)(rowBase + idx / I_KB) * I_KB + idx % I_KB];
    }
#pragma unroll
    for (int it = 0; it < 22; ++it) {
        int idx = tid + it * 256;
        int row = idx / 88, chunk = idx % 88;
        *(uint4*)(Alds + row * ASTRIDE2 + chunk * 16) =
            *(const uint4*)(iq + (size_t)(rowBase + row) * I_ + chunk * 16);
    }
    __syncthreads();

    int lane = tid & 31, wave = tid >> 5;
    int nIn = lane & 15, khalf = lane >> 4;
    int ncol = blockIdx.x * 128 + wave * 16 + nIn;
    const uint8_t* wp = wd + ((size_t)e * H_ + ncol) * I_;
    const float* sfD = d_sinv + ((size_t)e * H_NB + blockIdx.x) * I_KB;

    float acc[NSUB][8] = {{0}};
    for (int kb = 0; kb < I_KB; ++kb) {
        v16i b;
#pragma unroll
        for (int g = 0; g < 4; ++g) {
            uint4 q = *(const uint4*)(wp + kb * 128 + g * 32 + khalf * 16);
            b[4 * g] = (int)q.x; b[4 * g + 1] = (int)q.y;
            b[4 * g + 2] = (int)q.z; b[4 * g + 3] = (int)q.w;
        }
        float sb = sfD[kb];
#pragma unroll
        for (int s = 0; s < NSUB; ++s) {
            const uint8_t* arow = Alds + (size_t)(s * 16 + (lane & 15)) * ASTRIDE2;
            v16i a;
#pragma unroll
            for (int h = 0; h < 2; ++h)
#pragma unroll
                for (int p = 0; p < 4; ++p) {
                    uint2 d = *(const uint2*)(arow + kb * 128 + h * 64 + (p >> 1) * 32 + khalf * 16 + (p & 1) * 8);
                    a[8 * h + 2 * p]     = (int)d.x;
                    a[8 * h + 2 * p + 1] = (int)d.y;
                }
            v8f zero = {};
            v8f part = __builtin_amdgcn_wmma_f32_16x16x128_fp8_fp8(a, b, (short)0, zero, false, false);
            int mBase = s * 16 + khalf * 8;
#pragma unroll
            for (int r = 0; r < 8; ++r)
                acc[s][r] += part[r] * (isLds[(mBase + r) * I_KB + kb] * sb);
        }
    }
#pragma unroll
    for (int s = 0; s < NSUB; ++s) {
        int mBase = s * 16 + khalf * 8;
#pragma unroll
        for (int r = 0; r < 8; ++r) {
            int m = mBase + r;
            float wgt = wLds[m];
            if (wgt != 0.f)
                atomicAdd(&out[(size_t)tokLds[m] * H_ + ncol], acc[s][r] * wgt);
        }
    }
}

extern "C" void kernel_launch(void* const* d_in, const int* in_sizes, int n_in,
                              void* d_out, int out_size, void* d_ws, size_t ws_size,
                              hipStream_t stream) {
    const float*   hidden  = (const float*)d_in[0];
    const int*     tk_idx  = (const int*)d_in[1];
    const float*   tk_w    = (const float*)d_in[2];
    const uint8_t* w_gu    = (const uint8_t*)d_in[3];
    const float*   s_gu    = (const float*)d_in[4];
    const uint8_t* w_d     = (const uint8_t*)d_in[5];
    const float*   s_d     = (const float*)d_in[6];
    float*         out     = (float*)d_out;
    uint8_t*       ws      = (uint8_t*)d_ws;

    uint8_t* xq     = ws + OFF_XQ;
    float*   xs     = (float*)(ws + OFF_XS);
    uint8_t* iq     = ws + OFF_IQ;
    float*   isc    = (float*)(ws + OFF_IS);
    int*     rowTok = (int*)(ws + OFF_RTOK);
    float*   rowW   = (float*)(ws + OFF_RW);
    int*     cnt    = (int*)(ws + OFF_CNT);
    int*     cur    = (int*)(ws + OFF_CUR);
    int*     offs   = (int*)(ws + OFF_OFFS);
    int*     tileE  = (int*)(ws + OFF_TILEE);
    int*     tileR  = (int*)(ws + OFF_TILER);
    int*     nT     = (int*)(ws + OFF_NT);

    fe_init_route_k<<<(NROW + 255) / 256, 256, 0, stream>>>(rowTok, rowW, cnt, cur);
    fe_zero_out_k<<<(T_ * H_ / 4 + 255) / 256, 256, 0, stream>>>(out, T_ * H_ / 4);
    fe_act_quant_k<<<T_, 256, 0, stream>>>(hidden, xq, xs);
    fe_count_k<<<(T_ * TOPK_ + 255) / 256, 256, 0, stream>>>(tk_idx, cnt);
    fe_plan_k<<<1, 1, 0, stream>>>(cnt, offs, tileE, tileR, nT);
    fe_scatter_k<<<(T_ * TOPK_ + 255) / 256, 256, 0, stream>>>(tk_idx, tk_w, offs, cur, rowTok, rowW);

    dim3 g1(I_KB, NTILE_MAX);
    fe_gemm1_k<<<g1, 256, 0, stream>>>(xq, xs, w_gu, s_gu, rowTok, tileE, tileR, nT, iq, isc);
    dim3 g2(H_NB, NTILE_MAX);
    fe_gemm2_k<<<g2, 256, 0, stream>>>(iq, isc, w_d, s_d, rowTok, rowW, tileE, tileR, nT, out);
}
